// EdgeConv_5549097746953
// MI455X (gfx1250) — matrix operand order repaired and checked by reading, not compiled, not run on hardware
//
#include <hip/hip_runtime.h>

typedef __attribute__((ext_vector_type(16))) _Float16 v16h;
typedef __attribute__((ext_vector_type(8)))  _Float16 v8h;
typedef __attribute__((ext_vector_type(8)))  float    v8f;

#define NPTS  8192
#define BATCH 2
#define CIN   64
#define COUT  128
#define KNB   16
#define KNN_BLOCK 128

// total packed weight halves: Wn(8*512) + We(8*512) + Wm1(8*512) + Wm2(16*512)
#define WTOTAL_HALF (40 * 512)          /* 20480 halves = 40960 bytes */
#define WCHUNKS     (WTOTAL_HALF * 2 / 16) /* 2560 x 16B chunks */

__device__ __forceinline__ float leaky(float x) { return x >= 0.f ? x : 0.2f * x; }

// ---------------------------------------------------------------------------
// Kernel 1: brute-force kNN (top-16 by squared distance), one thread per query.
// Key = |p|^2 - 2<q,p>  (drops the constant |q|^2 term; same ordering).
// ---------------------------------------------------------------------------
__global__ __launch_bounds__(KNN_BLOCK)
void knn_kernel(const float* __restrict__ pos, int* __restrict__ idxb) {
    __shared__ float4 tile[KNN_BLOCK];
    const int q = blockIdx.x * KNN_BLOCK + threadIdx.x;   // b*NPTS + n
    const int b = q / NPTS;
    const int n = q - b * NPTS;
    const float* pb = pos + (size_t)b * NPTS * 3;
    const float qx = pb[n * 3 + 0], qy = pb[n * 3 + 1], qz = pb[n * 3 + 2];

    float bd[KNB];
    int   bi[KNB];
#pragma unroll
    for (int i = 0; i < KNB; ++i) { bd[i] = 3.0e38f; bi[i] = 0; }

    for (int t = 0; t < NPTS; t += KNN_BLOCK) {
        __syncthreads();
        const int p = t + threadIdx.x;
        const float x = pb[p * 3 + 0], y = pb[p * 3 + 1], z = pb[p * 3 + 2];
        tile[threadIdx.x] = make_float4(x, y, z, x * x + y * y + z * z);
        __syncthreads();
#pragma unroll 4
        for (int j = 0; j < KNN_BLOCK; ++j) {
            const float4 pt = tile[j];
            const float d = pt.w - 2.f * (qx * pt.x + qy * pt.y + qz * pt.z);
            if (d < bd[KNB - 1]) {                // candidate enters top-16
                float dd = d; int ci = t + j;     // bubble-insert (sorted ascending)
#pragma unroll
                for (int s = 0; s < KNB; ++s) {
                    const bool sw = dd < bd[s];
                    const float td = sw ? bd[s] : dd;
                    const int   ti = sw ? bi[s] : ci;
                    bd[s] = sw ? dd : bd[s];
                    bi[s] = sw ? ci : bi[s];
                    dd = td; ci = ti;
                }
            }
        }
    }
#pragma unroll
    for (int k = 0; k < KNB; ++k) idxb[q * KNB + k] = bi[k];
}

// ---------------------------------------------------------------------------
// Kernel 2: transpose feat [B, C, N] (f32) -> featT [B, N, C] (f16).
// ---------------------------------------------------------------------------
__global__ void transpose_feat(const float* __restrict__ feat,
                               _Float16* __restrict__ featT) {
    const int p = blockIdx.x * blockDim.x + threadIdx.x;  // b*NPTS + n
    if (p >= BATCH * NPTS) return;
    const int b = p / NPTS;
    const int n = p - b * NPTS;
    const float* src = feat + (size_t)b * CIN * NPTS + n;
    _Float16* dst = featT + (size_t)p * CIN;
#pragma unroll 8
    for (int c = 0; c < CIN; ++c)
        dst[c] = (_Float16)src[(size_t)c * NPTS];
}

// ---------------------------------------------------------------------------
// Kernel 3: pack a row-major weight [rows, 64] (f32) into a WMMA
// A-matrix f16 lane layout in contiguous k order (16x32 tiles). Tile = (mt, kk): rows 16mt..16mt+15,
// K = 32kk..32kk+31.  Lane l, element j holds:
//   row = 16*mt + (l&15),  K = 32*kk + j + (((l>>4)&1)<<4)
// ---------------------------------------------------------------------------
__global__ void pack_w(const float* __restrict__ W, _Float16* __restrict__ Wp,
                       int nTiles) {
    const int e = blockIdx.x * blockDim.x + threadIdx.x;
    if (e >= nTiles * 512) return;
    const int tile = e >> 9;
    const int lane = (e >> 4) & 31;
    const int j    = e & 15;
    const int mt = tile >> 1, kk = tile & 1;
    const int row = (mt << 4) + (lane & 15);
    const int K = (kk << 5) + j + (((lane >> 4) & 1) << 4);
    Wp[e] = (_Float16)W[row * CIN + K];
}

// ---------------------------------------------------------------------------
// Kernel 4: fused EdgeConv MLP. One wave per point (N dim of WMMA = 16
// neighbors). Weights staged block-wide into LDS with gfx1250 async
// global->LDS copies (ASYNCcnt), A-operands then come from ds_load_b128.
// 40 v_wmma_f32_16x16x32_f16 per wave; max-over-K = shfl_xor butterfly.
// ---------------------------------------------------------------------------
#define WMMA_F16(A, B, C) \
    __builtin_amdgcn_wmma_f32_16x16x32_f16(false, (A), false, (B), (short)0, (C), false, false)

__global__ __launch_bounds__(256)
void edgeconv_main(const _Float16* __restrict__ featT, const int* __restrict__ idxb,
                   const _Float16* __restrict__ Wall,
                   const float* __restrict__ bn, const float* __restrict__ be,
                   const float* __restrict__ bm1, const float* __restrict__ bm2,
                   float* __restrict__ out) {
    __shared__ __attribute__((aligned(32))) _Float16 wlds[WTOTAL_HALF]; // 40 KB
    __shared__ __attribute__((aligned(32))) _Float16 lds[8 * 1024];     // 16 KB
    const int lane = threadIdx.x & 31;
    const int wave = threadIdx.x >> 5;
    const int gw = blockIdx.x * 8 + wave;          // b*NPTS + n
    const int b = gw / NPTS;
    const int n = gw - b * NPTS;
    const int col = lane & 15;                     // neighbor column (WMMA N)
    const int hi  = (lane >> 4) & 1;
    _Float16* hb = lds + wave * 1024;              // 16 cols x 64 ch, [col][c]

    // --- stage all packed weights (40 KB) into LDS via async copy engine ---
    {
        const char* gsrc = (const char*)Wall;
        const unsigned ldst = (unsigned)(uintptr_t)(&wlds[0]);
#pragma unroll
        for (int i = 0; i < WCHUNKS / 256; ++i) {
            const int c = i * 256 + threadIdx.x;
            const void* g = gsrc + (size_t)c * 16;
            const unsigned d = ldst + c * 16;
            asm volatile("global_load_async_to_lds_b128 %0, %1, off"
                         :: "v"(d), "v"(g) : "memory");
        }
        asm volatile("s_wait_asynccnt 0" ::: "memory");
    }

    // --- gather B-matrices after the async copy has drained: g, e = g - center ---
    const int m = idxb[gw * KNB + col];
    const _Float16* src = featT + ((size_t)b * NPTS + m) * CIN;
    const _Float16* ctr = featT + (size_t)gw * CIN;
    const v16h g0 = *(const v16h*)(src + hi * 16);
    const v16h g1 = *(const v16h*)(src + 32 + hi * 16);
    const v16h c0 = *(const v16h*)(ctr + hi * 16);
    const v16h c1 = *(const v16h*)(ctr + 32 + hi * 16);
    const v16h e0 = g0 - c0;                       // packed f16 VALU
    const v16h e1 = g1 - c1;

    __syncthreads();                               // weights visible block-wide
    const _Float16* wn  = wlds;                    // 8 tiles
    const _Float16* we  = wlds + 4096;             // 8 tiles
    const _Float16* wm1 = wlds + 8192;             // 8 tiles
    const _Float16* wm2 = wlds + 12288;            // 16 tiles

    // --- layer 1: h1 = leaky(Wn*g + bn) + leaky(We*e + be)   [64 x 16]
#pragma unroll
    for (int mt = 0; mt < 4; ++mt) {
        v8f an = {}, ae = {};
        v16h A;
        A = *(const v16h*)(wn + (mt * 2 + 0) * 512 + lane * 16);  an = WMMA_F16(A, g0, an);
        A = *(const v16h*)(wn + (mt * 2 + 1) * 512 + lane * 16);  an = WMMA_F16(A, g1, an);
        A = *(const v16h*)(we + (mt * 2 + 0) * 512 + lane * 16);  ae = WMMA_F16(A, e0, ae);
        A = *(const v16h*)(we + (mt * 2 + 1) * 512 + lane * 16);  ae = WMMA_F16(A, e1, ae);
        const float* bnp = bn + mt * 16 + hi * 8;
        const float* bep = be + mt * 16 + hi * 8;
        v8h hh;
#pragma unroll
        for (int r = 0; r < 8; ++r)
            hh[r] = (_Float16)(leaky(an[r] + bnp[r]) + leaky(ae[r] + bep[r]));
        *(v8h*)(hb + col * 64 + mt * 16 + hi * 8) = hh;   // C-layout -> [col][c]
    }

    const v16h p0 = *(const v16h*)(hb + col * 64 + hi * 16);
    const v16h p1 = *(const v16h*)(hb + col * 64 + 32 + hi * 16);

    // --- layer 2: h2 = leaky(Wm1*h1 + bm1)   [64 x 16]
#pragma unroll
    for (int mt = 0; mt < 4; ++mt) {
        v8f a = {};
        v16h A;
        A = *(const v16h*)(wm1 + (mt * 2 + 0) * 512 + lane * 16); a = WMMA_F16(A, p0, a);
        A = *(const v16h*)(wm1 + (mt * 2 + 1) * 512 + lane * 16); a = WMMA_F16(A, p1, a);
        const float* bp = bm1 + mt * 16 + hi * 8;
        v8h hh;
#pragma unroll
        for (int r = 0; r < 8; ++r)
            hh[r] = (_Float16)leaky(a[r] + bp[r]);
        *(v8h*)(hb + col * 64 + mt * 16 + hi * 8) = hh;
    }

    const v16h q0 = *(const v16h*)(hb + col * 64 + hi * 16);
    const v16h q1 = *(const v16h*)(hb + col * 64 + 32 + hi * 16);

    // --- layer 3: h3 = leaky(Wm2*h2 + bm2)  [128 x 16], then max over K
#pragma unroll
    for (int mt = 0; mt < 8; ++mt) {
        v8f a = {};
        v16h A;
        A = *(const v16h*)(wm2 + (mt * 2 + 0) * 512 + lane * 16); a = WMMA_F16(A, q0, a);
        A = *(const v16h*)(wm2 + (mt * 2 + 1) * 512 + lane * 16); a = WMMA_F16(A, q1, a);
        const float* bp = bm2 + mt * 16 + hi * 8;
#pragma unroll
        for (int r = 0; r < 8; ++r) {
            float x = leaky(a[r] + bp[r]);
            // max over the 16 neighbor columns (lanes 0-15 / 16-31 butterflies)
            x = fmaxf(x, __shfl_xor(x, 1, 32));
            x = fmaxf(x, __shfl_xor(x, 2, 32));
            x = fmaxf(x, __shfl_xor(x, 4, 32));
            x = fmaxf(x, __shfl_xor(x, 8, 32));
            if (col == 0) {
                const int ch = mt * 16 + hi * 8 + r;
                out[((size_t)b * COUT + ch) * NPTS + n] = x;
            }
        }
    }
}

// ---------------------------------------------------------------------------
extern "C" void kernel_launch(void* const* d_in, const int* in_sizes, int n_in,
                              void* d_out, int out_size, void* d_ws, size_t ws_size,
                              hipStream_t stream) {
    (void)in_sizes; (void)n_in; (void)out_size; (void)ws_size;
    const float* feat   = (const float*)d_in[0];
    const float* pos    = (const float*)d_in[1];
    const float* W_node = (const float*)d_in[2];
    const float* b_node = (const float*)d_in[3];
    const float* W_edge = (const float*)d_in[4];
    const float* b_edge = (const float*)d_in[5];
    const float* W_m1   = (const float*)d_in[6];
    const float* b_m1   = (const float*)d_in[7];
    const float* W_m2   = (const float*)d_in[8];
    const float* b_m2   = (const float*)d_in[9];
    float* out = (float*)d_out;

    // workspace carve-out (all offsets 32B-aligned); weights contiguous so the
    // main kernel can async-copy them to LDS as one 40 KB region.
    char* ws = (char*)d_ws;
    int* idxb = (int*)ws;
    size_t off = (size_t)BATCH * NPTS * KNB * sizeof(int);          // 1 MB
    _Float16* featT = (_Float16*)(ws + off);
    off += (size_t)BATCH * NPTS * CIN * sizeof(_Float16);           // 2 MB
    _Float16* Wall = (_Float16*)(ws + off);
    _Float16* Wn  = Wall;
    _Float16* We  = Wall + 4096;
    _Float16* Wm1 = Wall + 8192;
    _Float16* Wm2 = Wall + 12288;

    knn_kernel<<<(BATCH * NPTS) / KNN_BLOCK, KNN_BLOCK, 0, stream>>>(pos, idxb);
    transpose_feat<<<(BATCH * NPTS + 255) / 256, 256, 0, stream>>>(feat, featT);
    pack_w<<<(8 * 512 + 255) / 256, 256, 0, stream>>>(W_node, Wn, 8);
    pack_w<<<(8 * 512 + 255) / 256, 256, 0, stream>>>(W_edge, We, 8);
    pack_w<<<(8 * 512 + 255) / 256, 256, 0, stream>>>(W_m1, Wm1, 8);
    pack_w<<<(16 * 512 + 255) / 256, 256, 0, stream>>>(W_m2, Wm2, 16);
    edgeconv_main<<<(BATCH * NPTS) / 8, 256, 0, stream>>>(
        featT, idxb, Wall, b_node, b_edge, b_m1, b_m2, out);
}
